// AttentionDecoder_61280593380086
// MI455X (gfx1250) — compile-verified, matrix operand order checked by reading, not run on hardware
//
#include <hip/hip_runtime.h>
#include <hip/hip_bf16.h>

// Problem constants (match reference)
#define BB 128
#define TT 20
#define VV 10000
#define EE 512
#define HH 1024
#define FF 196
#define DD 512

typedef __bf16 bf16_t;
typedef __attribute__((ext_vector_type(8)))  __bf16 v8bf;
typedef __attribute__((ext_vector_type(16))) __bf16 v16bf;
typedef __attribute__((ext_vector_type(8)))  float  v8f;

// ---------------------------------------------------------------------------
// Generic bf16 GEMM:  C[M,N] = A[M,K] * W[N,K]^T + bias[N]   (fp32 accumulate)
// Optional row mask: if lengths != nullptr, rows with lengths[row] < t are 0.
//
// Register-blocked: one wave computes a 16x64 strip (4 accumulators), so the
// A fragment is loaded once per k-step and reused for 4 WMMAs (2.5 loads per
// v_wmma_f32_16x16x32_bf16 instead of 4). Block = 128 threads (4 waves)
// covering 16 rows x 256 cols. Ragged N handled by clamping invalid tiles'
// weight pointers to row 0 (wave-uniform control flow; EXEC stays all-ones).
//
// A-fragment layout (16-bit A 16x32, ISA 7.12.2): lane<16 -> K {0..7,16..23},
// lane>=16 -> K {8..15,24..31}. W rows ([N,K] row-major == B^T) use the same
// pattern with N in place of M.
// ---------------------------------------------------------------------------
__global__ __launch_bounds__(128) void gemm_bf16_wmma(
    const bf16_t* __restrict__ A, int lda,
    const bf16_t* __restrict__ W, int ldw,
    const float* __restrict__ bias,
    float* __restrict__ C, long ldc,
    int M, int N, int K,
    const int* __restrict__ lengths, int t)
{
  const int wave   = threadIdx.x >> 5;
  const int lane   = threadIdx.x & 31;
  const int tile_m = blockIdx.y * 16;
  const int n_base = (blockIdx.x * 4 + wave) * 64;   // this wave's 16x64 strip
  if (tile_m >= M || n_base >= N) return;            // wave-uniform

  const int r    = lane & 15;
  const int half = lane >> 4;

  const bf16_t* arow = A + (long)(tile_m + r) * lda;

  bool valid[4];
  const bf16_t* wrow[4];
#pragma unroll
  for (int j = 0; j < 4; ++j) {
    int tn = n_base + j * 16;
    valid[j] = (tn < N);                      // N is always a multiple of 16
    wrow[j]  = W + (long)((valid[j] ? tn : 0) + r) * ldw;
  }

  v8f acc[4] = {};
  for (int k0 = 0; k0 < K; k0 += 32) {
    const int ka = k0 + half * 8;
    v8bf alo = *(const v8bf*)(arow + ka);
    v8bf ahi = *(const v8bf*)(arow + ka + 16);
    v16bf av = __builtin_shufflevector(alo, ahi, 0,1,2,3,4,5,6,7,8,9,10,11,12,13,14,15);
#pragma unroll
    for (int j = 0; j < 4; ++j) {
      v8bf blo = *(const v8bf*)(wrow[j] + ka);
      v8bf bhi = *(const v8bf*)(wrow[j] + ka + 16);
      v16bf bv = __builtin_shufflevector(blo, bhi, 0,1,2,3,4,5,6,7,8,9,10,11,12,13,14,15);
      acc[j] = __builtin_amdgcn_wmma_f32_16x16x32_bf16(
          /*neg_a=*/false, av, /*neg_b=*/false, bv,
          /*c_mod=*/(short)0, acc[j], /*reuse_a=*/false, /*reuse_b=*/false);
    }
  }

  // C/D layout: vgpr i, lanes 0-15 -> (M=i, N=lane); lanes 16-31 -> (M=i+8, N=lane-16)
#pragma unroll
  for (int j = 0; j < 4; ++j) {
    if (!valid[j]) continue;
    const int col = n_base + j * 16 + r;
    const float badd = bias ? bias[col] : 0.0f;
#pragma unroll
    for (int i = 0; i < 8; ++i) {
      int row = tile_m + i + (half << 3);
      float v = acc[j][i] + badd;
      if (lengths) v = (lengths[row] >= t) ? v : 0.0f;
      C[(long)row * ldc + col] = v;
    }
  }
}

// ---------------------------------------------------------------------------
// Helper kernels
// ---------------------------------------------------------------------------
__global__ void f32_to_bf16_kernel(const float* __restrict__ src,
                                   bf16_t* __restrict__ dst, long n) {
  long i = (long)blockIdx.x * blockDim.x + threadIdx.x;
  long stride = (long)gridDim.x * blockDim.x;
  for (; i < n; i += stride) dst[i] = (bf16_t)src[i];
}

__global__ void zero_u32_kernel(unsigned int* __restrict__ p, long n) {
  long i = (long)blockIdx.x * blockDim.x + threadIdx.x;
  long stride = (long)gridDim.x * blockDim.x;
  for (; i < n; i += stride) p[i] = 0u;
}

// Wcat[n, 0:1024] = W_ih[n,:], Wcat[n, 1024:2048] = W_hh[n,:]; bias_cat = b_ih + b_hh
__global__ void build_wcat_kernel(const float* __restrict__ W_ih,
                                  const float* __restrict__ W_hh,
                                  const float* __restrict__ b_ih,
                                  const float* __restrict__ b_hh,
                                  bf16_t* __restrict__ Wcat,
                                  float* __restrict__ bias_cat) {
  const long n = (long)(4 * HH) * 2048;
  long stride = (long)gridDim.x * blockDim.x;
  for (long i = (long)blockIdx.x * blockDim.x + threadIdx.x; i < n; i += stride) {
    long row = i >> 11;          // /2048
    long k   = i & 2047;
    float v = (k < 1024) ? W_ih[row * 1024 + k] : W_hh[row * 1024 + (k - 1024)];
    Wcat[i] = (bf16_t)v;
  }
  for (long j = (long)blockIdx.x * blockDim.x + threadIdx.x; j < 4 * HH; j += stride)
    bias_cat[j] = b_ih[j] + b_hh[j];
}

// z0 = mean(feature over F) written into xcat z-slot (used at t==0)
__global__ __launch_bounds__(256) void mean_feature_kernel(
    const float* __restrict__ feature, bf16_t* __restrict__ xcat_z) {
  int b = blockIdx.x;
  for (int d = threadIdx.x; d < DD; d += 256) {
    float s = 0.0f;
    const float* fp = feature + (long)b * FF * DD + d;
    for (int f = 0; f < FF; ++f) s += fp[(long)f * DD];
    xcat_z[(long)b * 2048 + d] = (bf16_t)(s * (1.0f / (float)FF));
  }
}

// xcat emb-slot[b, :] = emb[captions[b, t], :]
__global__ __launch_bounds__(256) void embed_kernel(
    const int* __restrict__ captions, const float* __restrict__ emb,
    bf16_t* __restrict__ xcat_e, int t) {
  int b = blockIdx.x;
  int cap = captions[b * TT + t];
  const float* e = emb + (long)cap * EE;
  for (int j = threadIdx.x; j < EE; j += 256)
    xcat_e[(long)b * 2048 + j] = (bf16_t)e[j];
}

// Attention: scores = relu(a_feat + a_h) . W_a2a + b_a2a ; softmax over F ;
// z = sum_f alpha_f * feature[:,f,:]  -> written (bf16) into xcat z-slot.
__global__ __launch_bounds__(256) void attention_kernel(
    const float* __restrict__ a_feat, const float* __restrict__ a_h,
    const float* __restrict__ feature, const float* __restrict__ W_a2a,
    const float* __restrict__ b_a2a, bf16_t* __restrict__ xcat_z) {
  __shared__ float sc[FF];
  __shared__ float red[256];
  const int b = blockIdx.x, tid = threadIdx.x;
  const int wave = tid >> 5, lane = tid & 31;
  const float* ah = a_h + (long)b * DD;

  for (int f = wave; f < FF; f += 8) {
    const float* af = a_feat + ((long)b * FF + f) * DD;
    float p = 0.0f;
    for (int d = lane; d < DD; d += 32) {
      float v = af[d] + ah[d];
      v = v > 0.0f ? v : 0.0f;
      p += v * W_a2a[d];
    }
    for (int off = 16; off > 0; off >>= 1) p += __shfl_xor(p, off, 32);
    if (lane == 0) sc[f] = p + b_a2a[0];
  }
  __syncthreads();

  // max
  float m = -3.0e38f;
  for (int f = tid; f < FF; f += 256) m = fmaxf(m, sc[f]);
  red[tid] = m; __syncthreads();
  for (int s = 128; s > 0; s >>= 1) {
    if (tid < s) red[tid] = fmaxf(red[tid], red[tid + s]);
    __syncthreads();
  }
  m = red[0]; __syncthreads();

  // exp + sum
  float ssum = 0.0f;
  for (int f = tid; f < FF; f += 256) {
    float e = __expf(sc[f] - m);
    sc[f] = e;
    ssum += e;
  }
  red[tid] = ssum; __syncthreads();
  for (int s = 128; s > 0; s >>= 1) {
    if (tid < s) red[tid] += red[tid + s];
    __syncthreads();
  }
  float inv = 1.0f / red[0]; __syncthreads();
  for (int f = tid; f < FF; f += 256) sc[f] *= inv;
  __syncthreads();

  // z = feature^T alpha
  for (int d = tid; d < DD; d += 256) {
    float acc = 0.0f;
    const float* fp = feature + (long)b * FF * DD + d;
    for (int f = 0; f < FF; ++f) acc += fp[(long)f * DD] * sc[f];
    xcat_z[(long)b * 2048 + d] = (bf16_t)acc;
  }
}

// LSTM cell: gates [B,4H] (order i,f,g,o). Updates masked h/c, writes masked h
// (bf16) into xcat h-slot, and unmasked h_new (bf16) for the FC GEMM.
__global__ __launch_bounds__(256) void lstm_kernel(
    const float* __restrict__ gates, float* __restrict__ h, float* __restrict__ c,
    const int* __restrict__ lengths, int t,
    bf16_t* __restrict__ xcat_h, bf16_t* __restrict__ hnew_bf) {
  const int b = blockIdx.x;
  const bool act = lengths[b] >= t;
  const long gb = (long)b * (4 * HH);
  for (int j = threadIdx.x; j < HH; j += 256) {
    float ig = 1.0f / (1.0f + __expf(-gates[gb + j]));
    float fg = 1.0f / (1.0f + __expf(-gates[gb + HH + j]));
    float gg = tanhf(gates[gb + 2 * HH + j]);
    float og = 1.0f / (1.0f + __expf(-gates[gb + 3 * HH + j]));
    long hi = (long)b * HH + j;
    float cn = fg * c[hi] + ig * gg;
    float hn = og * tanhf(cn);
    float hm = act ? hn : h[hi];
    float cm = act ? cn : c[hi];
    h[hi] = hm;
    c[hi] = cm;
    xcat_h[(long)b * 2048 + j] = (bf16_t)hm;
    hnew_bf[hi] = (bf16_t)hn;
  }
}

// ---------------------------------------------------------------------------
// Host orchestration
// ---------------------------------------------------------------------------
extern "C" void kernel_launch(void* const* d_in, const int* in_sizes, int n_in,
                              void* d_out, int out_size, void* d_ws, size_t ws_size,
                              hipStream_t stream) {
  (void)in_sizes; (void)n_in; (void)out_size; (void)ws_size;
  const float* feature = (const float*)d_in[0];
  const int*   captions= (const int*)  d_in[1];
  const int*   lengths = (const int*)  d_in[2];
  const float* emb     = (const float*)d_in[3];
  const float* W_ih    = (const float*)d_in[4];
  const float* b_ih    = (const float*)d_in[5];
  const float* W_hh    = (const float*)d_in[6];
  const float* b_hh    = (const float*)d_in[7];
  const float* W_fc    = (const float*)d_in[8];
  const float* b_fc    = (const float*)d_in[9];
  const float* W_h2a   = (const float*)d_in[10];
  const float* b_h2a   = (const float*)d_in[11];
  const float* W_f2a   = (const float*)d_in[12];
  const float* b_f2a   = (const float*)d_in[13];
  const float* W_a2a   = (const float*)d_in[14];
  const float* b_a2a   = (const float*)d_in[15];
  float* out = (float*)d_out;

  // carve workspace (256B-aligned slices)
  char* ws = (char*)d_ws;
  auto carve = [&](size_t bytes) -> char* {
    char* p = ws;
    ws += (bytes + 255) & ~(size_t)255;
    return p;
  };
  bf16_t* feat_bf  = (bf16_t*)carve((size_t)BB * FF * DD * 2);   // 25.7 MB
  bf16_t* Wf2a_bf  = (bf16_t*)carve((size_t)DD * DD * 2);
  bf16_t* Wh2a_bf  = (bf16_t*)carve((size_t)DD * HH * 2);
  bf16_t* Wfc_bf   = (bf16_t*)carve((size_t)VV * HH * 2);        // 20.5 MB
  bf16_t* Wcat_bf  = (bf16_t*)carve((size_t)(4 * HH) * 2048 * 2);// 16.8 MB
  float*  bias_cat = (float*) carve((size_t)(4 * HH) * 4);
  float*  a_feat   = (float*) carve((size_t)BB * FF * DD * 4);   // 51.4 MB
  float*  a_h      = (float*) carve((size_t)BB * DD * 4);
  bf16_t* xcat     = (bf16_t*)carve((size_t)BB * 2048 * 2);      // [z | emb | h]
  float*  gates    = (float*) carve((size_t)BB * 4 * HH * 4);
  float*  h_st     = (float*) carve((size_t)BB * HH * 4);
  float*  c_st     = (float*) carve((size_t)BB * HH * 4);
  bf16_t* hnew_bf  = (bf16_t*)carve((size_t)BB * HH * 2);

  bf16_t* xcat_z = xcat;           // cols 0..511
  bf16_t* xcat_e = xcat + EE;      // cols 512..1023
  bf16_t* xcat_h = xcat + 1024;    // cols 1024..2047

  auto gemm = [&](const bf16_t* A, int lda, const bf16_t* Wm, int ldw,
                  const float* bias, float* C, long ldc,
                  int M, int N, int K, const int* len, int t) {
    dim3 g((unsigned)((N + 255) / 256), (unsigned)((M + 15) / 16));
    gemm_bf16_wmma<<<g, dim3(128), 0, stream>>>(A, lda, Wm, ldw, bias, C, ldc,
                                                M, N, K, len, t);
  };

  // --- one-time (per call) preparation --------------------------------------
  {
    long n;
    n = (long)BB * FF * DD;
    f32_to_bf16_kernel<<<dim3(2048), dim3(256), 0, stream>>>(feature, feat_bf, n);
    n = (long)DD * DD;
    f32_to_bf16_kernel<<<dim3(256), dim3(256), 0, stream>>>(W_f2a, Wf2a_bf, n);
    n = (long)DD * HH;
    f32_to_bf16_kernel<<<dim3(512), dim3(256), 0, stream>>>(W_h2a, Wh2a_bf, n);
    n = (long)VV * HH;
    f32_to_bf16_kernel<<<dim3(2048), dim3(256), 0, stream>>>(W_fc, Wfc_bf, n);
    build_wcat_kernel<<<dim3(2048), dim3(256), 0, stream>>>(W_ih, W_hh, b_ih, b_hh,
                                                            Wcat_bf, bias_cat);
    zero_u32_kernel<<<dim3(256), dim3(256), 0, stream>>>((unsigned*)h_st, (long)BB * HH);
    zero_u32_kernel<<<dim3(256), dim3(256), 0, stream>>>((unsigned*)c_st, (long)BB * HH);
    zero_u32_kernel<<<dim3(256), dim3(256), 0, stream>>>((unsigned*)xcat, (long)BB * 2048 / 2);
    mean_feature_kernel<<<dim3(BB), dim3(256), 0, stream>>>(feature, xcat_z); // z0 for t=0
    // a_feat = feature @ W_f2a^T + b_f2a   [B*F, D]
    gemm(feat_bf, DD, Wf2a_bf, DD, b_f2a, a_feat, DD,
         BB * FF, DD, DD, nullptr, 0);
  }

  // --- recurrence -----------------------------------------------------------
  for (int t = 0; t < TT; ++t) {
    if (t > 0) {
      // a_h = h @ W_h2a^T + b_h2a  (A = masked h, bf16, from xcat h-slot)
      gemm(xcat_h, 2048, Wh2a_bf, HH, b_h2a, a_h, DD, BB, DD, HH, nullptr, 0);
      attention_kernel<<<dim3(BB), dim3(256), 0, stream>>>(a_feat, a_h, feature,
                                                           W_a2a, b_a2a, xcat_z);
    }
    embed_kernel<<<dim3(BB), dim3(256), 0, stream>>>(captions, emb, xcat_e, t);

    // gates = [z|emb|h] @ [W_ih|W_hh]^T + (b_ih + b_hh)   [B, 4H], K = 2048
    gemm(xcat, 2048, Wcat_bf, 2048, bias_cat, gates, 4 * HH,
         BB, 4 * HH, 2048, nullptr, 0);

    lstm_kernel<<<dim3(BB), dim3(256), 0, stream>>>(gates, h_st, c_st, lengths, t,
                                                    xcat_h, hnew_bf);

    // out[:, t, :] = (h_new @ W_fc^T + b_fc) * mask(lengths >= t)
    gemm(hnew_bf, HH, Wfc_bf, HH, b_fc, out + (long)t * VV, (long)TT * VV,
         BB, VV, HH, lengths, t);
  }
}
